// UnifiedModalEncoder_44719199486754
// MI455X (gfx1250) — matrix operand order repaired and checked by reading, not compiled, not run on hardware
//
#include <hip/hip_runtime.h>
#include <math.h>
#include <stdint.h>

// ---------------------------------------------------------------------------
// UnifiedModalEncoder forward for MI455X (gfx1250, wave32, WMMA).
// All GEMMs run through v_wmma_f32_16x16x32_bf16 with on-the-fly f32->bf16
// conversion during vectorized LDS staging (b128 loads, b64 LDS stores).
// ---------------------------------------------------------------------------

#define DIM   1024
#define HEADS 8
#define DH    128
#define NEXP  6
#define HIDD  4096
#define NPAT  196
#define PDIM  768
#define SLEN  77
#define BATCH 16

typedef __attribute__((ext_vector_type(16))) __bf16 v16bf;
typedef __attribute__((ext_vector_type(8)))  float  v8f;

union Frag16 { uint4 u[2]; v16bf v; };
union F4 { float4 v; float a[4]; };

__device__ __forceinline__ uint32_t bfbits(float f) {
  union { float f; uint32_t u; } a; a.f = f;
  return (a.u + 0x7FFFu + ((a.u >> 16) & 1u)) >> 16;   // RNE
}

__device__ __forceinline__ uint32_t pk2bf(float lo, float hi) {
#if __has_builtin(__builtin_amdgcn_cvt_pk_bf16_f32)
  return __builtin_bit_cast(uint32_t, __builtin_amdgcn_cvt_pk_bf16_f32(lo, hi));
#else
  return bfbits(lo) | (bfbits(hi) << 16);
#endif
}

// tanh-approx GELU rewritten as x*sigmoid(2z): one hardware exp, tiny code.
__device__ __forceinline__ float gelu_tanh(float x) {
  float u = 1.5957691216057308f * (x + 0.044715f * x * x * x);
  return x / (1.0f + __expf(-u));
}

// ---------------------------------------------------------------------------
// Generic GEMM: C[M,N] = act(A[M,K] @ B[K,N] + bias[N]); all row-major f32.
// Block: 256 threads (8 waves), tile 128(M) x 128(N), K staged 64 at a time
// (two k=32 WMMA sub-steps per barrier pair -> 16 WMMAs between barriers).
// Wave grid 4x2 -> each wave computes 32x64 = 8 wmma accumulators.
// Out-of-range M rows are CLAMPED during staging (never stored), keeping the
// staging loops branch-free and fully vectorized.
// ---------------------------------------------------------------------------
#define BM 128
#define BN 128
#define BK 64
#define LDA 72   // padded bf16 row stride (144B: 16B-aligned, spreads banks)
#define LDB 72

__global__ __launch_bounds__(256) void gemm_bf16_wmma(
    const float* __restrict__ A, const float* __restrict__ B,
    const float* __restrict__ bias, float* __restrict__ C,
    int M, int N, int K, int act)
{
  __shared__ __bf16 As[BM][LDA];   // [m][k]
  __shared__ __bf16 Bs[BN][LDB];   // transposed: [n][k]

  const int tid  = threadIdx.x;
  const int lane = tid & 31;
  const int wave = tid >> 5;
  const int wm   = wave >> 1;      // 0..3  (M direction, 32 rows each)
  const int wn   = wave & 1;       // 0..1  (N direction, 64 cols each)
  const int r    = lane & 15;
  const int hh   = lane >> 4;
  const int mBase = blockIdx.y * BM;
  const int nBase = blockIdx.x * BN;

  // B micro-tile ownership for staging: 8(K) x 4(N) per thread
  const int bk8 = (tid >> 5) << 3;        // 0,8,..,56  (K octet base)
  const int bn4 = (tid & 31) << 2;        // 0,4,..,124 (N quad base)

  v8f acc[2][4];
  #pragma unroll
  for (int i = 0; i < 2; ++i)
    #pragma unroll
    for (int j = 0; j < 4; ++j)
      #pragma unroll
      for (int e = 0; e < 8; ++e) acc[i][j][e] = 0.0f;

  for (int kk = 0; kk < K; kk += BK) {
    // speculative prefetch of next K tile (global_prefetch_b8; dropped if OOB)
    __builtin_prefetch(A + (size_t)(mBase + (tid >> 1)) * K + kk + BK, 0, 1);
    __builtin_prefetch(B + (size_t)(kk + BK + bk8) * N + nBase + bn4, 0, 1);

    // ---- stage A tile (128x64): 8 x float4 loads -> 8 x ds_store_b64
    #pragma unroll
    for (int i = 0; i < 8; ++i) {
      int c  = tid + i * 256;           // 2048 float4 chunks
      int m  = c >> 4;                  // 16 chunks per row
      int k4 = (c & 15) << 2;
      int gm = mBase + m; gm = (gm < M) ? gm : (M - 1);   // clamp, no branch
      F4 av; av.v = *(const float4*)(A + (size_t)gm * K + kk + k4);
      uint2 p; p.x = pk2bf(av.a[0], av.a[1]); p.y = pk2bf(av.a[2], av.a[3]);
      *(uint2*)&As[m][k4] = p;
    }
    // ---- stage B tile (64x128) transposed: 8 x float4 loads -> 8 x ds_store_b64
    {
      F4 br[8];
      #pragma unroll
      for (int i = 0; i < 8; ++i)
        br[i].v = *(const float4*)(B + (size_t)(kk + bk8 + i) * N + nBase + bn4);
      #pragma unroll
      for (int j = 0; j < 4; ++j) {     // column j -> contiguous K octet
        uint2 p0, p1;
        p0.x = pk2bf(br[0].a[j], br[1].a[j]);
        p0.y = pk2bf(br[2].a[j], br[3].a[j]);
        p1.x = pk2bf(br[4].a[j], br[5].a[j]);
        p1.y = pk2bf(br[6].a[j], br[7].a[j]);
        *(uint2*)&Bs[bn4 + j][bk8]     = p0;
        *(uint2*)&Bs[bn4 + j][bk8 + 4] = p1;
      }
    }
    __syncthreads();

    // two k=32 WMMA sub-steps per staged tile
    #pragma unroll
    for (int ks = 0; ks < BK; ks += 32) {
      // A fragment: elems 0..7 -> K = ks+hh*8.., elems 8..15 -> K = ks+16+hh*8..
      Frag16 af[2], bfr[4];
      #pragma unroll
      for (int i = 0; i < 2; ++i) {
        int row = wm * 32 + i * 16 + r;
        af[i].u[0] = *(const uint4*)&As[row][ks + hh * 8];
        af[i].u[1] = *(const uint4*)&As[row][ks + 16 + hh * 8];
      }
      // B fragment: elems 0..7 -> K = ks+hh*8.., elems 8..15 -> K = ks+16+hh*8..
      #pragma unroll
      for (int j = 0; j < 4; ++j) {
        int col = wn * 64 + j * 16 + r;
        bfr[j].u[0] = *(const uint4*)&Bs[col][ks + hh * 8];
        bfr[j].u[1] = *(const uint4*)&Bs[col][ks + 16 + hh * 8];
      }
      #pragma unroll
      for (int i = 0; i < 2; ++i)
        #pragma unroll
        for (int j = 0; j < 4; ++j)
          acc[i][j] = __builtin_amdgcn_wmma_f32_16x16x32_bf16(
              false, af[i].v, false, bfr[j].v, (short)0, acc[i][j], false, false);
    }
    __syncthreads();
  }

  // epilogue: C VGPR e -> M = hh*8 + e, N = lane&15 (bias always present)
  const bool fullM = (mBase + BM) <= M;
  #pragma unroll
  for (int i = 0; i < 2; ++i) {
    #pragma unroll
    for (int j = 0; j < 4; ++j) {
      int col = nBase + wn * 64 + j * 16 + r;
      float bv = bias[col];
      int rowBase = mBase + wm * 32 + i * 16 + hh * 8;
      if (fullM) {
        #pragma unroll
        for (int e = 0; e < 8; ++e) {
          float v = acc[i][j][e] + bv;
          if (act == 1) v = gelu_tanh(v);
          C[(size_t)(rowBase + e) * N + col] = v;
        }
      } else {
        #pragma unroll
        for (int e = 0; e < 8; ++e) {
          if (rowBase + e < M) {
            float v = acc[i][j][e] + bv;
            if (act == 1) v = gelu_tanh(v);
            C[(size_t)(rowBase + e) * N + col] = v;
          }
        }
      }
    }
  }
}

// ---------------------------------------------------------------------------
// Fused attention: one block per (b, h, q). q/k/v layout: row t=b*S+s, col h*DH+d.
// ---------------------------------------------------------------------------
__global__ __launch_bounds__(128) void attn_kernel(
    const float* __restrict__ q, const float* __restrict__ k,
    const float* __restrict__ v, float* __restrict__ o, int S)
{
  __shared__ float qsh[DH];
  __shared__ float sc[224];
  __shared__ float red[128];
  const int tid = threadIdx.x;
  const int id  = blockIdx.x;
  const int sq  = id % S;
  const int bh  = id / S;
  const int h   = bh % HEADS;
  const int b   = bh / HEADS;
  const size_t qoff = (size_t)(b * S + sq) * DIM + h * DH;
  qsh[tid] = q[qoff + tid];
  __syncthreads();
  const float scale = 0.08838834764831845f;  // 1/sqrt(128)
  for (int s = tid; s < S; s += 128) {
    const float* kr = k + (size_t)(b * S + s) * DIM + h * DH;
    float dot = 0.0f;
    #pragma unroll 4
    for (int d0 = 0; d0 < DH; d0 += 4) {
      F4 kv; kv.v = *(const float4*)(kr + d0);
      dot += qsh[d0] * kv.a[0] + qsh[d0+1] * kv.a[1]
           + qsh[d0+2] * kv.a[2] + qsh[d0+3] * kv.a[3];
    }
    sc[s] = dot * scale;
  }
  __syncthreads();
  float m = -1e30f;
  for (int s = tid; s < S; s += 128) m = fmaxf(m, sc[s]);
  red[tid] = m; __syncthreads();
  for (int off = 64; off > 0; off >>= 1) {
    if (tid < off) red[tid] = fmaxf(red[tid], red[tid + off]);
    __syncthreads();
  }
  float mx = red[0]; __syncthreads();
  float p = 0.0f;
  for (int s = tid; s < S; s += 128) {
    float e = __expf(sc[s] - mx); sc[s] = e; p += e;
  }
  red[tid] = p; __syncthreads();
  for (int off = 64; off > 0; off >>= 1) {
    if (tid < off) red[tid] += red[tid + off];
    __syncthreads();
  }
  float denom = red[0];
  float accv = 0.0f;
  for (int s = 0; s < S; ++s)
    accv += sc[s] * v[(size_t)(b * S + s) * DIM + h * DH + tid];
  o[qoff + tid] = accv / denom;
}

// ---------------------------------------------------------------------------
__global__ __launch_bounds__(256) void ln_kernel(
    const float* __restrict__ x, const float* __restrict__ g,
    const float* __restrict__ b, float* __restrict__ y)
{
  __shared__ float r1[256], r2[256];
  const int t = blockIdx.x, tid = threadIdx.x;
  const float* xr = x + (size_t)t * DIM;
  // D = 1024 = 256 threads * 1 float4 each
  F4 xv; xv.v = *(const float4*)(xr + tid * 4);
  float s  = xv.a[0] + xv.a[1] + xv.a[2] + xv.a[3];
  float s2 = xv.a[0]*xv.a[0] + xv.a[1]*xv.a[1] + xv.a[2]*xv.a[2] + xv.a[3]*xv.a[3];
  r1[tid] = s; r2[tid] = s2; __syncthreads();
  for (int off = 128; off > 0; off >>= 1) {
    if (tid < off) { r1[tid] += r1[tid + off]; r2[tid] += r2[tid + off]; }
    __syncthreads();
  }
  float mean = r1[0] * (1.0f / DIM);
  float var  = r2[0] * (1.0f / DIM) - mean * mean;
  float inv  = rsqrtf(var + 1e-5f);
  F4 gv; gv.v = *(const float4*)(g + tid * 4);
  F4 bv; bv.v = *(const float4*)(b + tid * 4);
  F4 ov;
  #pragma unroll
  for (int e = 0; e < 4; ++e) ov.a[e] = (xv.a[e] - mean) * inv * gv.a[e] + bv.a[e];
  *(float4*)(y + (size_t)t * DIM + tid * 4) = ov.v;
}

// router: logits -> softmax -> top-2 mask; atomics for load-balance loss stats
__global__ __launch_bounds__(64) void router_kernel(
    const float* __restrict__ x, const float* __restrict__ rW,
    const float* __restrict__ rb, float* __restrict__ masks,
    float* __restrict__ stats, int T, int E)
{
  __shared__ float red[64][8];
  const int t = blockIdx.x, tid = threadIdx.x;
  float part[8];
  #pragma unroll
  for (int e = 0; e < 8; ++e) part[e] = 0.0f;
  for (int d0 = tid; d0 < DIM; d0 += 64) {
    float xv = x[(size_t)t * DIM + d0];
    for (int e = 0; e < E; ++e) part[e] += xv * rW[d0 * E + e];
  }
  #pragma unroll
  for (int e = 0; e < 8; ++e) red[tid][e] = part[e];
  __syncthreads();
  for (int off = 32; off > 0; off >>= 1) {
    if (tid < off)
      for (int e = 0; e < 8; ++e) red[tid][e] += red[tid + off][e];
    __syncthreads();
  }
  if (tid == 0) {
    float lg[8], mx = -1e30f;
    for (int e = 0; e < E; ++e) { lg[e] = red[0][e] + rb[e]; mx = fmaxf(mx, lg[e]); }
    float sum = 0.0f;
    for (int e = 0; e < E; ++e) { lg[e] = __expf(lg[e] - mx); sum += lg[e]; }
    int i1 = 0;
    for (int e = 1; e < E; ++e) if (lg[e] > lg[i1]) i1 = e;
    int i2 = (i1 == 0) ? 1 : 0;
    for (int e = 0; e < E; ++e) if (e != i1 && lg[e] > lg[i2]) i2 = e;
    for (int e = 0; e < E; ++e) {
      float pr = lg[e] / sum;
      masks[(size_t)t * E + e] = (e == i1 || e == i2) ? pr : 0.0f;
      atomicAdd(&stats[e], pr);            // me accumulation
    }
    atomicAdd(&stats[8 + i1], 1.0f);       // ce accumulation
    atomicAdd(&stats[8 + i2], 1.0f);
  }
}

__global__ void scale_add_kernel(float* __restrict__ out, const float* __restrict__ y,
                                 const float* __restrict__ masks, int E, int e,
                                 int n, int accum)
{
  int idx = blockIdx.x * 256 + threadIdx.x;
  if (idx >= n) return;
  int t = idx / DIM;
  float v = masks[(size_t)t * E + e] * y[idx];
  if (accum) out[idx] += v; else out[idx] = v;
}

__global__ void add_kernel(float* __restrict__ a, const float* __restrict__ b, int n) {
  int idx = blockIdx.x * 256 + threadIdx.x;
  if (idx < n) a[idx] += b[idx];
}

__global__ void addpos_kernel(float* __restrict__ x, const float* __restrict__ pos,
                              int S, int n) {
  int idx = blockIdx.x * 256 + threadIdx.x;
  if (idx >= n) return;
  int d0 = idx % DIM, s = (idx / DIM) % S;
  x[idx] += pos[(size_t)s * DIM + d0];
}

__global__ void embed_kernel(const int* __restrict__ ids, const float* __restrict__ emb,
                             const float* __restrict__ pos, float* __restrict__ x, int n) {
  int idx = blockIdx.x * 256 + threadIdx.x;
  if (idx >= n) return;
  int d0 = idx % DIM, t = idx / DIM, s = t % SLEN;
  x[idx] = emb[(size_t)ids[t] * DIM + d0] + pos[(size_t)s * DIM + d0];
}

__global__ void patchify_kernel(const float* __restrict__ img, float* __restrict__ patches,
                                int n) {
  int idx = blockIdx.x * 256 + threadIdx.x;
  if (idx >= n) return;
  int b  = idx / (NPAT * PDIM);
  int r0 = idx % (NPAT * PDIM);
  int np = r0 / PDIM, pd = r0 % PDIM;
  int hh = np / 14, ww = np % 14;
  int p1 = pd / 48, r2 = pd % 48, p2 = r2 / 3, c = r2 % 3;
  patches[idx] = img[(((size_t)b * 3 + c) * 224 + hh * 16 + p1) * 224 + ww * 16 + p2];
}

__global__ void mean_kernel(const float* __restrict__ x, float* __restrict__ f,
                            int S, int n) {
  int idx = blockIdx.x * 256 + threadIdx.x;
  if (idx >= n) return;
  int b = idx / DIM, d0 = idx % DIM;
  float s = 0.0f;
  for (int i = 0; i < S; ++i) s += x[(size_t)(b * S + i) * DIM + d0];
  f[idx] = s / (float)S;
}

__global__ void comb_kernel(const float* __restrict__ fi, const float* __restrict__ ft,
                            const float* __restrict__ mod, float* __restrict__ comb, int n) {
  int idx = blockIdx.x * 256 + threadIdx.x;
  if (idx >= n) return;
  int d0 = idx % DIM, r0 = (idx / DIM) & 1, b = idx / (2 * DIM);
  comb[idx] = ((r0 == 0) ? fi[(size_t)b * DIM + d0] : ft[(size_t)b * DIM + d0])
              + mod[(size_t)r0 * DIM + d0];
}

__global__ void zero_kernel(float* p, int n) {
  int idx = blockIdx.x * blockDim.x + threadIdx.x;
  if (idx < n) p[idx] = 0.0f;
}

__global__ void finalize_kernel(const float* __restrict__ stats, float* __restrict__ out) {
  if (blockIdx.x == 0 && threadIdx.x == 0) {
    const float Ti = 16.0f * 196.0f, Tt = 16.0f * 77.0f, Tu = 32.0f;
    float loss = 0.0f, l;
    l = 0.0f; for (int e = 0; e < 7; ++e) l += (stats[e]      / Ti) * (stats[8 + e]  / Ti);
    loss += 7.0f * l;
    l = 0.0f; for (int e = 0; e < 7; ++e) l += (stats[16 + e] / Tt) * (stats[24 + e] / Tt);
    loss += 7.0f * l;
    l = 0.0f; for (int e = 0; e < 8; ++e) l += (stats[32 + e] / Tu) * (stats[40 + e] / Tu);
    loss += 8.0f * l;
    out[32768] = loss;
  }
}

// ---------------------------------------------------------------------------
// Host orchestration
// ---------------------------------------------------------------------------
// d_in layout: setup_inputs() insertion order:
//   0: image, 1: input_ids, then params in construction order.
enum { E_Wq = 0, E_bq, E_Wk, E_bk, E_Wv, E_bv, E_Wo, E_bo,
       E_ln1g, E_ln1b, E_ln2g, E_ln2b, E_ln3g, E_ln3b,
       E_eW1, E_eb1, E_eW2, E_eb2, E_sW1, E_sb1, E_sW2, E_sb2, E_rW, E_rb,
       E_COUNT };
// globals after two encoder blocks (base = 2):
//   50 img_pW, 51 img_pb, 52 img_pos, 53 emb, 54 txt_pos, 55 mod,
//   56..59 uni_eW1/eb1/eW2/eb2, 60..63 uni_i*, 64..67 uni_t*, 68 uni_rW, 69 uni_rb

extern "C" void kernel_launch(void* const* d_in, const int* in_sizes, int n_in,
                              void* d_out, int out_size, void* d_ws, size_t ws_size,
                              hipStream_t stream)
{
  (void)in_sizes; (void)n_in; (void)out_size; (void)ws_size;
  auto F = [&](int i) { return (const float*)d_in[i]; };
  const float* image = F(0);
  const int*   ids   = (const int*)d_in[1];
  float*       out   = (float*)d_out;

  const int Timg = BATCH * NPAT;   // 3136
  const int Ttxt = BATCH * SLEN;   // 1232
  const int Tu   = 2 * BATCH;      // 32

  // workspace carve (~165 MB of f32)
  float* w = (float*)d_ws;
  auto alloc = [&](size_t nf) { float* p = w; w += nf; return p; };
  float* patches = alloc((size_t)Timg * PDIM);
  float* xb   = alloc((size_t)Timg * DIM);
  float* lnb  = alloc((size_t)Timg * DIM);
  float* qb   = alloc((size_t)Timg * DIM);
  float* kb   = alloc((size_t)Timg * DIM);
  float* vb   = alloc((size_t)Timg * DIM);
  float* ab   = alloc((size_t)Timg * DIM);
  float* hb   = alloc((size_t)Timg * HIDD);
  float* yb   = alloc((size_t)Timg * DIM);
  float* eob  = alloc((size_t)Timg * DIM);
  float* mkb  = alloc((size_t)Timg * 8);
  float* fimg = alloc((size_t)BATCH * DIM);
  float* ftxt = alloc((size_t)BATCH * DIM);
  float* comb = alloc((size_t)Tu * DIM);
  float* stats = alloc(48);

  auto gemm = [&](const float* A, const float* B, const float* bias, float* C,
                  int M, int N, int K, int act) {
    dim3 g(N / BN, (M + BM - 1) / BM);
    gemm_bf16_wmma<<<g, 256, 0, stream>>>(A, B, bias, C, M, N, K, act);
  };
  auto ew = [&](int n) { return dim3((n + 255) / 256); };

  zero_kernel<<<1, 64, 0, stream>>>(stats, 48);

  // ---- image patch embedding
  patchify_kernel<<<ew(Timg * PDIM), 256, 0, stream>>>(image, patches, Timg * PDIM);
  gemm(patches, F(50), F(51), xb, Timg, DIM, PDIM, 0);
  addpos_kernel<<<ew(Timg * DIM), 256, 0, stream>>>(xb, F(52), NPAT, Timg * DIM);

  auto encoder = [&](int base, int S, float* fout, float* statp) {
    int T = BATCH * S, n = T * DIM;
    ln_kernel<<<T, 256, 0, stream>>>(xb, F(base + E_ln1g), F(base + E_ln1b), lnb);
    gemm(lnb, F(base + E_Wq), F(base + E_bq), qb, T, DIM, DIM, 0);
    gemm(lnb, F(base + E_Wk), F(base + E_bk), kb, T, DIM, DIM, 0);
    gemm(lnb, F(base + E_Wv), F(base + E_bv), vb, T, DIM, DIM, 0);
    attn_kernel<<<BATCH * HEADS * S, 128, 0, stream>>>(qb, kb, vb, ab, S);
    gemm(ab, F(base + E_Wo), F(base + E_bo), yb, T, DIM, DIM, 0);
    add_kernel<<<ew(n), 256, 0, stream>>>(xb, yb, n);
    ln_kernel<<<T, 256, 0, stream>>>(xb, F(base + E_ln2g), F(base + E_ln2b), lnb);
    router_kernel<<<T, 64, 0, stream>>>(lnb, F(base + E_rW), F(base + E_rb),
                                        mkb, statp, T, NEXP + 1);
    for (int e = 0; e < NEXP; ++e) {
      gemm(xb, F(base + E_eW1) + (size_t)e * DIM * HIDD,
           F(base + E_eb1) + (size_t)e * HIDD, hb, T, HIDD, DIM, 1);
      gemm(hb, F(base + E_eW2) + (size_t)e * HIDD * DIM,
           F(base + E_eb2) + (size_t)e * DIM, yb, T, DIM, HIDD, 0);
      scale_add_kernel<<<ew(n), 256, 0, stream>>>(eob, yb, mkb, NEXP + 1, e, n, e > 0);
    }
    gemm(xb, F(base + E_sW1), F(base + E_sb1), hb, T, HIDD, DIM, 1);
    gemm(hb, F(base + E_sW2), F(base + E_sb2), yb, T, DIM, HIDD, 0);
    scale_add_kernel<<<ew(n), 256, 0, stream>>>(eob, yb, mkb, NEXP + 1, NEXP, n, 1);
    ln_kernel<<<T, 256, 0, stream>>>(eob, F(base + E_ln3g), F(base + E_ln3b), lnb);
    mean_kernel<<<ew(BATCH * DIM), 256, 0, stream>>>(lnb, fout, S, BATCH * DIM);
  };

  encoder(2, NPAT, fimg, stats);                 // image encoder
  embed_kernel<<<ew(Ttxt * DIM), 256, 0, stream>>>(ids, F(53), F(54), xb, Ttxt * DIM);
  encoder(2 + E_COUNT, SLEN, ftxt, stats + 16);  // text encoder

  // ---- unified MoE head (writes directly into d_out)
  comb_kernel<<<ew(Tu * DIM), 256, 0, stream>>>(fimg, ftxt, F(55), comb, Tu * DIM);
  router_kernel<<<Tu, 64, 0, stream>>>(comb, F(68), F(69), mkb, stats + 32, Tu, NEXP + 2);
  for (int e = 0; e < NEXP; ++e) {
    gemm(comb, F(56) + (size_t)e * DIM * HIDD, F(57) + (size_t)e * HIDD, hb, Tu, HIDD, DIM, 1);
    gemm(hb, F(58) + (size_t)e * HIDD * DIM, F(59) + (size_t)e * DIM, yb, Tu, DIM, HIDD, 0);
    scale_add_kernel<<<ew(Tu * DIM), 256, 0, stream>>>(out, yb, mkb, NEXP + 2, e, Tu * DIM, e > 0);
  }
  gemm(comb, F(60), F(61), hb, Tu, HIDD, DIM, 1);
  gemm(hb, F(62), F(63), yb, Tu, DIM, HIDD, 0);
  scale_add_kernel<<<ew(Tu * DIM), 256, 0, stream>>>(out, yb, mkb, NEXP + 2, NEXP, Tu * DIM, 1);
  gemm(comb, F(64), F(65), hb, Tu, HIDD, DIM, 1);
  gemm(hb, F(66), F(67), yb, Tu, DIM, HIDD, 0);
  scale_add_kernel<<<ew(Tu * DIM), 256, 0, stream>>>(out, yb, mkb, NEXP + 2, NEXP + 1, Tu * DIM, 1);

  finalize_kernel<<<1, 1, 0, stream>>>(stats, out);
}
